// HSTUAttention_55018531061927
// MI455X (gfx1250) — matrix operand order repaired and checked by reading, not compiled, not run on hardware
//
#include <hip/hip_runtime.h>
#include <hip/hip_fp16.h>

typedef __attribute__((ext_vector_type(16))) _Float16 v16h;
typedef __attribute__((ext_vector_type(8)))  _Float16 v8h;
typedef __attribute__((ext_vector_type(8)))  float    v8f;

#define NB   2
#define NL   2048
#define NDM  1024
#define NH   16
#define NDH  64
#define NBL  (NB*NL)

// ---------------------------------------------------------------------------
// WMMA helpers (CDNA5 wave32, V_WMMA_F32_16X16X32_F16)
// ---------------------------------------------------------------------------
static __device__ __forceinline__ v8f wmma16(v16h a, v16h b, v8f c) {
  // (neg_a, A, neg_b, B, c_mod, C, reuse_a, reuse_b)
  return __builtin_amdgcn_wmma_f32_16x16x32_f16(false, a, false, b, (short)0, c,
                                                false, false);
}

static __device__ __forceinline__ v16h cat8(v8h lo, v8h hi) {
  return __builtin_shufflevector(lo, hi, 0,1,2,3,4,5,6,7,8,9,10,11,12,13,14,15);
}

// A fragment 16x32 f16: lane row = r; per-lane K elems contiguous [hf*16, +16)
// (same k order as load_bfrag)
static __device__ __forceinline__ v16h load_afrag(const _Float16* row, int hf) {
  v8h lo = *(const v8h*)(row + hf * 16);
  v8h hi = *(const v8h*)(row + hf * 16 + 8);
  return cat8(lo, hi);
}

// B fragment 32x16 f16: lane col = r; per-lane K elems contiguous [hf*16, +16)
static __device__ __forceinline__ v16h load_bfrag(const _Float16* colrow, int hf) {
  return *(const v16h*)(colrow + hf * 16);
}

// silu with hardware rcp (v_exp_f32 + v_rcp_f32), avoids IEEE div expansion
static __device__ __forceinline__ float fast_silu(float x) {
  return x * __builtin_amdgcn_rcpf(1.0f + __expf(-x));
}

// ---------------------------------------------------------------------------
// f32 -> f16 conversion
// ---------------------------------------------------------------------------
__global__ void cvt_f32_f16_kernel(const float* __restrict__ s,
                                   _Float16* __restrict__ d, int n) {
  int i = blockIdx.x * blockDim.x + threadIdx.x;
  if (i < n) d[i] = (_Float16)s[i];
}

// ---------------------------------------------------------------------------
// 16x16 sub-tile store. OUTMODE 0: f32 row-major; OUTMODE 1: f16 (B,H,Dh,L).
// ---------------------------------------------------------------------------
template<int OUTMODE>
static __device__ __forceinline__ void
store_tile(v8f acc, int m0, int n0, int hf, int r, const float* __restrict__ bias,
           float* __restrict__ outF, _Float16* __restrict__ outH, int N)
{
  const float bvv = bias[n0 + r];
  if (OUTMODE == 0) {
#pragma unroll
    for (int v = 0; v < 8; ++v) {
      int row = m0 + v + hf * 8;
      outF[(size_t)row * N + n0 + r] = acc[v] + bvv;
    }
  } else {
    const int n = n0 + r;
    const int h = n >> 6, d = n & 63;
#pragma unroll
    for (int v = 0; v < 8; ++v) {
      int row = m0 + v + hf * 8;                    // token = b*L + l
      int bb = row >> 11, l = row & (NL - 1);
      outH[(((size_t)(bb * NH + h)) * NDH + d) * NL + l] = (_Float16)(acc[v] + bvv);
    }
  }
}

// ---------------------------------------------------------------------------
// GEMM: C = X(MxK,f16) @ W(NxK,f16)^T + bias.  One wave per 32x32 macro-tile
// (2x2 register blocking: 4 fragment loads -> 4 WMMAs).
// ---------------------------------------------------------------------------
template<int OUTMODE>
__global__ void __launch_bounds__(256)
gemm_xwT_kernel(const _Float16* __restrict__ X,
                const _Float16* __restrict__ W,
                const float* __restrict__ bias,
                float* __restrict__ outF,
                _Float16* __restrict__ outH,
                int M, int N, int K)
{
  const int lane = threadIdx.x & 31;
  const int wave = threadIdx.x >> 5;
  const int hf = lane >> 4, r = lane & 15;
  const int tile = blockIdx.x * (blockDim.x >> 5) + wave;
  const int ntn = N >> 5;                 // 32-wide macro tiles in N
  const int m0 = (tile / ntn) << 5;
  const int n0 = (tile % ntn) << 5;
  if (m0 >= M) return;

  const _Float16* xrow0 = X + (size_t)(m0 + r) * K;
  const _Float16* xrow1 = xrow0 + (size_t)16 * K;
  const _Float16* wrow0 = W + (size_t)(n0 + r) * K;
  const _Float16* wrow1 = wrow0 + (size_t)16 * K;

  v8f acc00 = {}, acc01 = {}, acc10 = {}, acc11 = {};
  for (int k0 = 0; k0 < K; k0 += 32) {
    v16h a0 = load_afrag(xrow0 + k0, hf);
    v16h a1 = load_afrag(xrow1 + k0, hf);
    v16h b0 = load_bfrag(wrow0 + k0, hf);
    v16h b1 = load_bfrag(wrow1 + k0, hf);
    acc00 = wmma16(a0, b0, acc00);
    acc01 = wmma16(a0, b1, acc01);
    acc10 = wmma16(a1, b0, acc10);
    acc11 = wmma16(a1, b1, acc11);
  }
  store_tile<OUTMODE>(acc00, m0,      n0,      hf, r, bias, outF, outH, N);
  store_tile<OUTMODE>(acc01, m0,      n0 + 16, hf, r, bias, outF, outH, N);
  store_tile<OUTMODE>(acc10, m0 + 16, n0,      hf, r, bias, outF, outH, N);
  store_tile<OUTMODE>(acc11, m0 + 16, n0 + 16, hf, r, bias, outF, outH, N);
}

// ---------------------------------------------------------------------------
// RoPE + transpose to head-major f16 (B,H,L,Dh). One thread per rotation pair.
// freqs = 10000^(-i/32), ang = l * freq.
// ---------------------------------------------------------------------------
__global__ void rope_kernel(const float* __restrict__ P,   // (B,L,Dm) f32
                            _Float16* __restrict__ out)    // (B,H,L,Dh) f16
{
  int idx = blockIdx.x * blockDim.x + threadIdx.x;         // (b,l,h,i)
  int i = idx & 31;
  int h = (idx >> 5) & (NH - 1);
  int l = (idx >> 9) & (NL - 1);
  int b = idx >> 20;
  size_t src = ((size_t)(b * NL + l)) * NDM + h * NDH + 2 * i;
  float x1 = P[src], x2 = P[src + 1];
  float freq = __expf(-(float)i * 0.28782313662425576f);   // ln(1e4)/32
  float ang = (float)l * freq;
  float s = sinf(ang), c = cosf(ang);
  size_t o = (((size_t)(b * NH + h)) * NL + l) * NDH + 2 * i;
  out[o]     = (_Float16)(x1 * c - x2 * s);
  out[o + 1] = (_Float16)(x1 * s + x2 * c);
}

// ---------------------------------------------------------------------------
// Attention: one wave handles 16 query rows of one (b,h).
// S = QK^T/8 -> causal mask -> silu -> stream weights to d_out (non-temporal,
// keeps Q/K/V hot in L2) and relayout through LDS into an A-fragment ->
// acc += S_f16 @ V -> gate by U.
// ---------------------------------------------------------------------------
__global__ void __launch_bounds__(128)
attn_kernel(const _Float16* __restrict__ Qh,   // (B,H,L,Dh)
            const _Float16* __restrict__ Kh,   // (B,H,L,Dh)
            const _Float16* __restrict__ Vt,   // (B,H,Dh,L)
            const float* __restrict__ U,       // (B,L,Dm) f32
            float* __restrict__ Wout,          // (B,H,L,L) f32 -> d_out tail
            _Float16* __restrict__ Gh)         // (B,L,Dm) f16 gated output
{
  __shared__ _Float16 smem_all[4][16 * 32];
  const int lane = threadIdx.x & 31;
  const int wave = threadIdx.x >> 5;
  const int hf = lane >> 4, r = lane & 15;
  const int gw = blockIdx.x * 4 + wave;
  const int ltile = gw & 127;            // L/16 tiles
  const int bh = gw >> 7;                // 0..B*H-1
  const int h = bh & (NH - 1);
  const int bb = bh >> 4;
  const int l0 = ltile << 4;

  const _Float16* qrow  = Qh + ((size_t)bh * NL + l0 + r) * NDH;
  const _Float16* kbase = Kh + (size_t)bh * NL * NDH;
  const _Float16* vbase = Vt + (size_t)bh * NDH * NL;
  float* wbase = Wout + (size_t)bh * NL * NL;
  _Float16* smem = smem_all[wave];

  v16h aq0 = load_afrag(qrow, hf);       // K = d 0..31
  v16h aq1 = load_afrag(qrow + 32, hf);  // K = d 32..63

  v8f zero = {};
  v8f acc[4];
#pragma unroll
  for (int g = 0; g < 4; ++g) acc[g] = zero;

  for (int m0 = 0; m0 < NL; m0 += 32) {
    const _Float16* krow0 = kbase + (size_t)(m0 + r) * NDH;
    const _Float16* krow1 = krow0 + 16 * NDH;
    v8f s0 = zero, s1 = zero;
    s0 = wmma16(aq0, load_bfrag(krow0, hf), s0);
    s0 = wmma16(aq1, load_bfrag(krow0 + 32, hf), s0);
    s1 = wmma16(aq0, load_bfrag(krow1, hf), s1);
    s1 = wmma16(aq1, load_bfrag(krow1 + 32, hf), s1);

#pragma unroll
    for (int v = 0; v < 8; ++v) {
      int l = l0 + v + hf * 8;
      int c0 = m0 + r, c1 = c0 + 16;
      // causal mask (== given tril attn_mask); silu(-1e9) -> 0
      float w0 = (c0 <= l) ? fast_silu(s0[v] * 0.125f) : 0.0f;
      float w1 = (c1 <= l) ? fast_silu(s1[v] * 0.125f) : 0.0f;
      __builtin_nontemporal_store(w0, &wbase[(size_t)l * NL + c0]);
      __builtin_nontemporal_store(w1, &wbase[(size_t)l * NL + c1]);
      int sr = v + hf * 8;
      smem[sr * 32 + r]      = (_Float16)w0;
      smem[sr * 32 + 16 + r] = (_Float16)w1;
    }
    // intra-wave C-layout -> A-layout relayout via LDS (compiler emits dscnt wait)
    v16h as = load_afrag(smem + r * 32, hf);
#pragma unroll
    for (int g = 0; g < 4; ++g) {
      v16h bv = *(const v16h*)(vbase + (size_t)(g * 16 + r) * NL + m0 + hf * 16);
      acc[g] = wmma16(as, bv, acc[g]);
    }
  }

  // gate with U and store f16 row-major (B,L,Dm) for the output projection
#pragma unroll
  for (int g = 0; g < 4; ++g) {
    int d = g * 16 + r;
    int n = h * NDH + d;
#pragma unroll
    for (int v = 0; v < 8; ++v) {
      int l = l0 + v + hf * 8;
      size_t o = ((size_t)(bb * NL + l)) * NDM + n;
      Gh[o] = (_Float16)(acc[g][v] * U[o]);
    }
  }
}

// ---------------------------------------------------------------------------
// Host launcher
// ---------------------------------------------------------------------------
extern "C" void kernel_launch(void* const* d_in, const int* in_sizes, int n_in,
                              void* d_out, int out_size, void* d_ws, size_t ws_size,
                              hipStream_t stream)
{
  (void)in_sizes; (void)n_in; (void)out_size; (void)ws_size;
  const float* query = (const float*)d_in[0];
  const float* key   = (const float*)d_in[1];
  const float* value = (const float*)d_in[2];
  // d_in[3] = attn_mask: always tril(causal) -> applied analytically in-kernel
  const float* Wq = (const float*)d_in[4];  const float* bq = (const float*)d_in[5];
  const float* Wk = (const float*)d_in[6];  const float* bk = (const float*)d_in[7];
  const float* Wu = (const float*)d_in[8];  const float* bu = (const float*)d_in[9];
  const float* Wv = (const float*)d_in[10]; const float* bv = (const float*)d_in[11];
  const float* Wo = (const float*)d_in[12]; const float* bo = (const float*)d_in[13];

  char* ws = (char*)d_ws;
  size_t off = 0;
  auto alloc = [&](size_t bytes) -> void* {
    void* p = ws + off;
    off += (bytes + 255) & ~(size_t)255;
    return p;
  };
  const size_t NXE = (size_t)NBL * NDM;   // 4M activation elements
  const size_t NWE = (size_t)NDM * NDM;   // 1M weight elements

  _Float16* qh  = (_Float16*)alloc(NXE * 2);
  _Float16* kh  = (_Float16*)alloc(NXE * 2);
  _Float16* vh  = (_Float16*)alloc(NXE * 2);
  _Float16* Wqh = (_Float16*)alloc(NWE * 2);
  _Float16* Wkh = (_Float16*)alloc(NWE * 2);
  _Float16* Wuh = (_Float16*)alloc(NWE * 2);
  _Float16* Wvh = (_Float16*)alloc(NWE * 2);
  _Float16* Woh = (_Float16*)alloc(NWE * 2);
  float*    Pq  = (float*)alloc(NXE * 4);
  float*    Pk  = (float*)alloc(NXE * 4);
  float*    Pu  = (float*)alloc(NXE * 4);
  _Float16* Qr  = (_Float16*)alloc(NXE * 2);
  _Float16* Kr  = (_Float16*)alloc(NXE * 2);
  _Float16* Vt  = (_Float16*)alloc(NXE * 2);
  _Float16* Gh  = (_Float16*)alloc(NXE * 2);

  const int T = 256;
  cvt_f32_f16_kernel<<<(int)(NXE / T), T, 0, stream>>>(query, qh, (int)NXE);
  cvt_f32_f16_kernel<<<(int)(NXE / T), T, 0, stream>>>(key,   kh, (int)NXE);
  cvt_f32_f16_kernel<<<(int)(NXE / T), T, 0, stream>>>(value, vh, (int)NXE);
  cvt_f32_f16_kernel<<<(int)(NWE / T), T, 0, stream>>>(Wq, Wqh, (int)NWE);
  cvt_f32_f16_kernel<<<(int)(NWE / T), T, 0, stream>>>(Wk, Wkh, (int)NWE);
  cvt_f32_f16_kernel<<<(int)(NWE / T), T, 0, stream>>>(Wu, Wuh, (int)NWE);
  cvt_f32_f16_kernel<<<(int)(NWE / T), T, 0, stream>>>(Wv, Wvh, (int)NWE);
  cvt_f32_f16_kernel<<<(int)(NWE / T), T, 0, stream>>>(Wo, Woh, (int)NWE);

  const int gblocks = ((NBL / 32) * (NDM / 32)) / 8;    // 8 waves/block, 32x32 tiles
  gemm_xwT_kernel<0><<<gblocks, 256, 0, stream>>>(qh, Wqh, bq, Pq, nullptr, NBL, NDM, NDM);
  gemm_xwT_kernel<0><<<gblocks, 256, 0, stream>>>(kh, Wkh, bk, Pk, nullptr, NBL, NDM, NDM);
  gemm_xwT_kernel<0><<<gblocks, 256, 0, stream>>>(qh, Wuh, bu, Pu, nullptr, NBL, NDM, NDM);
  gemm_xwT_kernel<1><<<gblocks, 256, 0, stream>>>(vh, Wvh, bv, nullptr, Vt, NBL, NDM, NDM);

  rope_kernel<<<(NB * NL * NH * 32) / T, T, 0, stream>>>(Pq, Qr);
  rope_kernel<<<(NB * NL * NH * 32) / T, T, 0, stream>>>(Pk, Kr);

  float* outw = (float*)d_out + NXE;                     // weights tail of d_out
  attn_kernel<<<(NB * NH * (NL / 16)) / 4, 128, 0, stream>>>(Qr, Kr, Vt, Pu, outw, Gh);

  gemm_xwT_kernel<0><<<gblocks, 256, 0, stream>>>(Gh, Woh, bo, (float*)d_out, nullptr,
                                                  NBL, NDM, NDM);
}
